// MultiHeadAttention_90512140796090
// MI455X (gfx1250) — matrix operand order repaired and checked by reading, not compiled, not run on hardware
//
#include <hip/hip_runtime.h>
#include <hip/hip_bf16.h>

// ---------------------------------------------------------------------------
// MHA forward for MI455X (gfx1250, wave32, WMMA).
//   x:[2,2048,1024] f32, w_q/w_k/w_v/w_o:[1024,1024] f32 (nn.Linear: y = x @ W^T)
//   out:[2,2048,1024] f32
// Pipeline (bf16 WMMA, f32 accumulate):
//   1) cvt f32->bf16 (x + 4 weights)
//   2) Q = (x @ Wq^T) * 1/8 (bf16), K = x@Wk^T, V = x@Wv^T (bf16) — LDS-staged W
//   3) causal flash-attention per head -> ctx (bf16) — LDS-staged K/V
//   4) out = ctx @ Wo^T (f32)
// K/W staging uses GLOBAL_LOAD_ASYNC_TO_LDS_B128 (ASYNCcnt) when available.
// ---------------------------------------------------------------------------

#define D_MODEL 1024
#define N_HEADS 16
#define HEAD_DIM 64
#define SEQ 2048
#define BATCH 2
#define M_TOK (BATCH * SEQ) /* 4096 */

typedef __bf16 bf16_t;
typedef __attribute__((ext_vector_type(16))) __bf16 v16bf;
typedef __attribute__((ext_vector_type(8)))  __bf16 v8bf;
typedef __attribute__((ext_vector_type(8)))  float   v8f;
typedef __attribute__((ext_vector_type(4)))  int     v4i;

#if defined(__has_builtin)
#if __has_builtin(__builtin_amdgcn_global_load_async_to_lds_b128) && \
    __has_builtin(__builtin_amdgcn_s_wait_asynccnt)
#define HAVE_ASYNC_LDS 1
#endif
#endif

#ifdef HAVE_ASYNC_LDS
// Builtin prototype (per hipcc diagnostic): (v4i addrspace(1)*, v4i addrspace(3)*, Ii, Ii)
typedef __attribute__((address_space(1))) v4i* g_v4i_p;
typedef __attribute__((address_space(3))) v4i* l_v4i_p;
static __device__ __forceinline__ void async_b128(const bf16_t* g, bf16_t* l) {
  __builtin_amdgcn_global_load_async_to_lds_b128(
      (g_v4i_p)(void*)const_cast<bf16_t*>(g), (l_v4i_p)(void*)l, 0, 0);
}
#endif

static __device__ __forceinline__ v16bf cat8(v8bf lo, v8bf hi) {
  return __builtin_shufflevector(lo, hi, 0,1,2,3,4,5,6,7,8,9,10,11,12,13,14,15);
}
static __device__ __forceinline__ v8bf ld8(const bf16_t* p) {
  return *reinterpret_cast<const v8bf*>(p);
}
static __device__ __forceinline__ v16bf ld16p(const bf16_t* p) {
  // 16 contiguous bf16 as two 16B-aligned halves (safe with padded LDS strides)
  return cat8(ld8(p), ld8(p + 8));
}
static __device__ __forceinline__ v8f wmma_bf16(v16bf a, v16bf b, v8f c) {
  // D = A(16x32) * B(32x16) + C ; f32 accumulate
  return __builtin_amdgcn_wmma_f32_16x16x32_bf16(false, a, false, b, (short)0, c,
                                                 false, false);
}

// ---------------------------------------------------------------------------
// 1) f32 -> bf16 conversion (grid-stride)
// ---------------------------------------------------------------------------
__global__ void cvt_f32_bf16(const float* __restrict__ src,
                             bf16_t* __restrict__ dst, int n) {
  int i = blockIdx.x * blockDim.x + threadIdx.x;
  int stride = gridDim.x * blockDim.x;
  for (; i < n; i += stride) dst[i] = (bf16_t)src[i];
}

// ---------------------------------------------------------------------------
// 2/4) Y[M,N] = A[M,K] @ W[N,K]^T   (K = N = 1024, M = 4096)
// Block: 256 threads = 8 waves; block tile 128(M) x 64(N); wave tile 16x64.
// W tile (64 rows x 64 K) staged in LDS once per block (8x reuse across waves).
// ---------------------------------------------------------------------------
#define WSTR 72  /* padded LDS row stride (elements) to spread banks */

template <bool OUT_BF16>
__global__ __launch_bounds__(256) void gemm_xwT(const bf16_t* __restrict__ A,
                                                const bf16_t* __restrict__ W,
                                                void* __restrict__ Yv,
                                                float scale) {
  __shared__ __align__(32) bf16_t Wl[64 * WSTR];

  const int lane = threadIdx.x & 31;
  const int wave = threadIdx.x >> 5;
  const int m16  = lane & 15;   // row (A-frag) / col (B-frag,D) within tile
  const int kh   = lane >> 4;   // K-half selector per ISA layout
  const int row0 = blockIdx.y * 128 + wave * 16;
  const int col0 = blockIdx.x * 64;

  v8f acc[4] = {};
  const bf16_t* arow = A + (size_t)(row0 + m16) * D_MODEL;

  for (int kb = 0; kb < D_MODEL; kb += 64) {
    // ---- stage W[col0..col0+63][kb..kb+63] into LDS (whole block) ----
    __syncthreads();
    {
      const int nloc = threadIdx.x >> 2;       // 0..63
      const int k0   = (threadIdx.x & 3) * 16; // 0,16,32,48
      const bf16_t* src = W + (size_t)(col0 + nloc) * D_MODEL + kb + k0;
      bf16_t* dst = &Wl[nloc * WSTR + k0];
#ifdef HAVE_ASYNC_LDS
      async_b128(src, dst);
      async_b128(src + 8, dst + 8);
#else
      *reinterpret_cast<v8bf*>(dst)     = ld8(src);
      *reinterpret_cast<v8bf*>(dst + 8) = ld8(src + 8);
#endif
    }
#ifdef HAVE_ASYNC_LDS
    __builtin_amdgcn_s_wait_asynccnt(0);
#endif
    __syncthreads();

#pragma unroll
    for (int ks = 0; ks < 2; ++ks) {
      const int k = kb + ks * 32;
      // A 16x32 frag: lane<16 -> K {k..k+7, k+16..k+23}; lane>=16 shifted by 8
      v16bf af = cat8(ld8(arow + k + kh * 8), ld8(arow + k + 16 + kh * 8));
#pragma unroll
      for (int nt = 0; nt < 4; ++nt) {
        // B 32x16 frag: lane holds column n = staged W row (nt*16+m16)
        v16bf bf = cat8(ld8(&Wl[(nt * 16 + m16) * WSTR + ks * 32 + kh * 8]),
                        ld8(&Wl[(nt * 16 + m16) * WSTR + ks * 32 + 16 + kh * 8]));
        acc[nt] = wmma_bf16(af, bf, acc[nt]);
      }
    }
  }

#pragma unroll
  for (int nt = 0; nt < 4; ++nt) {
    const int col = col0 + nt * 16 + m16;
#pragma unroll
    for (int v = 0; v < 8; ++v) {
      const int row = row0 + v + 8 * kh;  // D layout: VGPR v -> row v + 8*lane[4]
      const float val = acc[nt][v] * scale;
      if (OUT_BF16)
        ((bf16_t*)Yv)[(size_t)row * D_MODEL + col] = (bf16_t)val;
      else
        ((float*)Yv)[(size_t)row * D_MODEL + col] = val;
    }
  }
}

// ---------------------------------------------------------------------------
// 3) Causal flash attention.
// Grid: x = 16 query blocks of 128, y = 32 (batch*head).
// Block: 8 waves; wave = 16 queries. K/V staged in LDS in 32-key blocks.
// Q carries the 1/sqrt(Dh) scale already.
// ---------------------------------------------------------------------------
#define KSTR 72  /* Kl padded row stride */
#define VSTR 40  /* VT padded row stride */
#define PSTR 40  /* Pl padded row stride */

__global__ __launch_bounds__(256) void flash_attn(const bf16_t* __restrict__ Q,
                                                  const bf16_t* __restrict__ Kc,
                                                  const bf16_t* __restrict__ Vc,
                                                  bf16_t* __restrict__ O) {
  __shared__ __align__(32) bf16_t Kl[32 * KSTR];     // K block, [key][d]
  __shared__ __align__(32) bf16_t VT[64 * VSTR];     // V block transposed [d][key]
  __shared__ __align__(32) bf16_t Pl[8][16 * PSTR];  // per-wave P staging

  const int lane = threadIdx.x & 31;
  const int wave = threadIdx.x >> 5;
  const int m16  = lane & 15;
  const int kh   = lane >> 4;
  const int bh   = blockIdx.y;
  const int b    = bh >> 4;
  const int h    = bh & 15;
  const size_t tok0 = (size_t)b * SEQ;
  const int q0   = blockIdx.x * 128 + wave * 16;  // wave's query tile (seq idx)

  // Q fragments for d=[0,32) and [32,64), resident all kernel
  const bf16_t* qptr = Q + (tok0 + q0 + m16) * D_MODEL + h * HEAD_DIM;
  v16bf qf[2];
#pragma unroll
  for (int dh = 0; dh < 2; ++dh)
    qf[dh] = cat8(ld8(qptr + dh * 32 + kh * 8),
                  ld8(qptr + dh * 32 + 16 + kh * 8));

  v8f   o[4] = {};
  float mrun[8], lrun[8];
#pragma unroll
  for (int v = 0; v < 8; ++v) { mrun[v] = -__builtin_inff(); lrun[v] = 0.f; }

  const int nKB = blockIdx.x * 4 + 4;  // 32-wide key blocks covering causal span
  for (int kbI = 0; kbI < nKB; ++kbI) {
    const int kbase = kbI * 32;

    // ---- stage K/V block (whole workgroup, convergent) ----
    __syncthreads();
    {
      const int kk = threadIdx.x >> 3;       // 0..31 key within block
      const int d0 = (threadIdx.x & 7) * 8;  // 0..56
      const size_t gk = (tok0 + kbase + kk) * D_MODEL + h * HEAD_DIM + d0;
#ifdef HAVE_ASYNC_LDS
      async_b128(Kc + gk, &Kl[kk * KSTR + d0]);  // no VGPR round trip
#else
      *reinterpret_cast<v8bf*>(&Kl[kk * KSTR + d0]) = ld8(Kc + gk);
#endif
      v8bf vv = ld8(Vc + gk);                    // V needs transpose via VGPRs
#pragma unroll
      for (int i = 0; i < 8; ++i) VT[(d0 + i) * VSTR + kk] = vv[i];
      if (kbI + 1 < nKB) {  // prefetch next block (global_prefetch_b8)
        __builtin_prefetch(Kc + gk + 32 * D_MODEL, 0, 1);
        __builtin_prefetch(Vc + gk + 32 * D_MODEL, 0, 1);
      }
    }
#ifdef HAVE_ASYNC_LDS
    __builtin_amdgcn_s_wait_asynccnt(0);
#endif
    __syncthreads();

    if (kbase > q0 + 15) continue;  // block fully above diagonal for this wave

    // ---- scores: Q(16x64) @ K^T(64x32) as 2 key sub-tiles x 2 d-steps ----
    v8f s[2] = {};
#pragma unroll
    for (int t = 0; t < 2; ++t) {
#pragma unroll
      for (int dh = 0; dh < 2; ++dh) {
        v16bf bf = cat8(ld8(&Kl[(t * 16 + m16) * KSTR + dh * 32 + kh * 8]),
                        ld8(&Kl[(t * 16 + m16) * KSTR + dh * 32 + 16 + kh * 8]));
        s[t] = wmma_bf16(qf[dh], bf, s[t]);
      }
    }

    // ---- causal mask (only partially-masked blocks) ----
    if (kbase + 31 > q0) {
#pragma unroll
      for (int t = 0; t < 2; ++t) {
        const int keyc = kbase + t * 16 + m16;
#pragma unroll
        for (int v = 0; v < 8; ++v)
          if (keyc > q0 + v + 8 * kh) s[t][v] = -__builtin_inff();
      }
    }

    // ---- online softmax (row reductions across 16-lane halves, wave32) ----
    float p0[8], p1[8];
#pragma unroll
    for (int v = 0; v < 8; ++v) {
      float rm = fmaxf(s[0][v], s[1][v]);
      rm = fmaxf(rm, __shfl_xor(rm, 1));
      rm = fmaxf(rm, __shfl_xor(rm, 2));
      rm = fmaxf(rm, __shfl_xor(rm, 4));
      rm = fmaxf(rm, __shfl_xor(rm, 8));
      const float mnew = fmaxf(mrun[v], rm);
      const float corr = __expf(mrun[v] - mnew);
      const float e0 = __expf(s[0][v] - mnew);
      const float e1 = __expf(s[1][v] - mnew);
      float rs = e0 + e1;
      rs += __shfl_xor(rs, 1);
      rs += __shfl_xor(rs, 2);
      rs += __shfl_xor(rs, 4);
      rs += __shfl_xor(rs, 8);
      lrun[v] = lrun[v] * corr + rs;
      mrun[v] = mnew;
      p0[v] = e0;
      p1[v] = e1;
#pragma unroll
      for (int nt = 0; nt < 4; ++nt) o[nt][v] *= corr;
    }

    // ---- P: D-layout -> A-layout via per-wave LDS round trip ----
#pragma unroll
    for (int v = 0; v < 8; ++v) {
      const int r = v + 8 * kh;
      Pl[wave][r * PSTR + m16]      = (bf16_t)p0[v];
      Pl[wave][r * PSTR + 16 + m16] = (bf16_t)p1[v];
    }
    __builtin_amdgcn_wave_barrier();  // keep LDS store->load order (same wave)
    v16bf pA = cat8(ld8(&Pl[wave][m16 * PSTR + kh * 8]),
                    ld8(&Pl[wave][m16 * PSTR + 16 + kh * 8]));

    // ---- O += P(16x32) @ V(32x64), 4 N-tiles over Dh ----
#pragma unroll
    for (int nt = 0; nt < 4; ++nt) {
      v16bf vB = cat8(ld8(&VT[(nt * 16 + m16) * VSTR + kh * 8]),
                      ld8(&VT[(nt * 16 + m16) * VSTR + 16 + kh * 8]));
      o[nt] = wmma_bf16(pA, vB, o[nt]);
    }
  }

  // ---- normalize + store context (bf16) ----
#pragma unroll
  for (int nt = 0; nt < 4; ++nt) {
    const int col = h * HEAD_DIM + nt * 16 + m16;
#pragma unroll
    for (int v = 0; v < 8; ++v) {
      const int r = v + 8 * kh;
      O[(tok0 + q0 + r) * D_MODEL + col] = (bf16_t)(o[nt][v] / lrun[v]);
    }
  }
}

// ---------------------------------------------------------------------------
extern "C" void kernel_launch(void* const* d_in, const int* in_sizes, int n_in,
                              void* d_out, int out_size, void* d_ws,
                              size_t ws_size, hipStream_t stream) {
  const float* x  = (const float*)d_in[0];
  const float* wq = (const float*)d_in[1];
  const float* wk = (const float*)d_in[2];
  const float* wv = (const float*)d_in[3];
  const float* wo = (const float*)d_in[4];

  char* ws = (char*)d_ws;
  size_t off = 0;
  auto take = [&](size_t bytes) -> char* {
    char* p = ws + off;
    off += (bytes + 255) & ~(size_t)255;
    return p;
  };
  const size_t XB = (size_t)M_TOK * D_MODEL * sizeof(bf16_t);   // 8 MB
  const size_t WB = (size_t)D_MODEL * D_MODEL * sizeof(bf16_t); // 2 MB
  bf16_t* xb  = (bf16_t*)take(XB);
  bf16_t* wqb = (bf16_t*)take(WB);
  bf16_t* wkb = (bf16_t*)take(WB);
  bf16_t* wvb = (bf16_t*)take(WB);
  bf16_t* wob = (bf16_t*)take(WB);
  bf16_t* Qb  = (bf16_t*)take(XB);
  bf16_t* Kb  = (bf16_t*)take(XB);
  bf16_t* Vb  = (bf16_t*)take(XB);
  bf16_t* Cb  = (bf16_t*)take(XB);
  (void)ws_size; (void)in_sizes; (void)n_in; (void)out_size;

  // 1) convert to bf16
  cvt_f32_bf16<<<2048, 256, 0, stream>>>(x,  xb,  M_TOK * D_MODEL);
  cvt_f32_bf16<<<1024, 256, 0, stream>>>(wq, wqb, D_MODEL * D_MODEL);
  cvt_f32_bf16<<<1024, 256, 0, stream>>>(wk, wkb, D_MODEL * D_MODEL);
  cvt_f32_bf16<<<1024, 256, 0, stream>>>(wv, wvb, D_MODEL * D_MODEL);
  cvt_f32_bf16<<<1024, 256, 0, stream>>>(wo, wob, D_MODEL * D_MODEL);

  // 2) projections (softmax scale folded into Q)
  dim3 ggrid(D_MODEL / 64, M_TOK / 128);  // (16, 32)
  gemm_xwT<true><<<ggrid, 256, 0, stream>>>(xb, wqb, Qb, 0.125f);
  gemm_xwT<true><<<ggrid, 256, 0, stream>>>(xb, wkb, Kb, 1.0f);
  gemm_xwT<true><<<ggrid, 256, 0, stream>>>(xb, wvb, Vb, 1.0f);

  // 3) causal flash attention
  dim3 agrid(SEQ / 128, BATCH * N_HEADS);  // (16, 32)
  flash_attn<<<agrid, 256, 0, stream>>>(Qb, Kb, Vb, Cb);

  // 4) output projection -> f32
  gemm_xwT<false><<<ggrid, 256, 0, stream>>>(Cb, wob, (float*)d_out, 1.0f);
}
